// CustomInteractionBlock_64450279244117
// MI455X (gfx1250) — matrix operand order repaired and checked by reading, not compiled, not run on hardware
//
#include <hip/hip_runtime.h>
#include <math.h>

// CustomInteractionBlock for MI455X (gfx1250, wave32, WMMA).
//
// Factorized form: out_p[e,w] = sum_{c,u} h[e,c] * t_p[e,u] * Wfc2[c,p,u,w]
//   => GEMM  M=E, K=2048 (c*32+u), N=32 with A built on the fly as an outer
//      product (never materialized in HBM) and B repacked from W_fc2 into
//      per-lane WMMA fragment order. Heavy math on v_wmma_f32_16x16x32_bf16;
//      B streamed block-wide through LDS with global_load_async_to_lds_b128
//      (ASYNCcnt double-buffering) so all 4 waves share one L2 fetch.

#define MULC 32

typedef __attribute__((ext_vector_type(16))) __bf16 v16bf;
typedef __attribute__((ext_vector_type(8)))  float  v8f;

__device__ __forceinline__ float bf2f(unsigned short b) {
  union { unsigned int u; float f; } v; v.u = ((unsigned int)b) << 16; return v.f;
}
__device__ __forceinline__ unsigned short f2bf(float f) {
  union { float f; unsigned int u; } v; v.f = f;
  unsigned int r = v.u + 0x7FFFu + ((v.u >> 16) & 1u);  // round-to-nearest-even
  return (unsigned short)(r >> 16);
}

// ---------------------------------------------------------------------------
// Kernel 1: radial basis + fc1 + silu  ->  h[E][64] in bf16
// ---------------------------------------------------------------------------
__global__ void k_radial_h(const float* __restrict__ edge_length,
                           const float* __restrict__ W_fc1,
                           unsigned short* __restrict__ Hbuf, int E) {
  int idx = blockIdx.x * blockDim.x + threadIdx.x;
  if (idx >= E * 64) return;
  int e = idx >> 6, c = idx & 63;
  float el = edge_length[e];
  const float sigma  = 5.0f / 7.0f;                 // 5/(NRAD-1)
  const float inv2s2 = 1.0f / (2.0f * sigma * sigma);
  float acc = 0.f;
#pragma unroll
  for (int r = 0; r < 8; ++r) {
    float d = el - (5.0f / 7.0f) * (float)r;        // linspace(0,5,8)
    float rad = __expf(-d * d * inv2s2);
    acc += rad * W_fc1[r * 64 + c];
  }
  acc *= 0.35355339059327373f;                      // 1/sqrt(NRAD)
  float h = acc / (1.0f + __expf(-acc));            // silu
  Hbuf[idx] = f2bf(h);
}

// ---------------------------------------------------------------------------
// Kernel 2: tensor-product "t" vectors -> Tbuf[E][11][32] bf16
//   rows: 0:t00  1:t11  2..4:t01_k  5..7:t10_k  8..10:t12_k
// ---------------------------------------------------------------------------
__global__ void k_tvec(const float* __restrict__ x,
                       const float* __restrict__ edge_attr,
                       const int* __restrict__ edge_src,
                       unsigned short* __restrict__ Tbuf, int E) {
  int idx = blockIdx.x * blockDim.x + threadIdx.x;
  if (idx >= E * 32) return;
  int e = idx >> 5, u = idx & 31;
  const float* ea = edge_attr + (size_t)e * 9;
  float y0 = ea[0];
  float y1_0 = ea[1], y1_1 = ea[2], y1_2 = ea[3];
  float y2_0 = ea[4], y2_1 = ea[5], y2_2 = ea[6], y2_3 = ea[7], y2_4 = ea[8];
  int src = edge_src[e];
  const float* xr = x + (size_t)src * 128;
  float sj  = xr[u];
  float vj0 = xr[32 + u * 3 + 0];
  float vj1 = xr[32 + u * 3 + 1];
  float vj2 = xr[32 + u * 3 + 2];

  const float IS3 = 0.5773502691896258f;   // 1/sqrt(3)
  const float s2  = 0.7071067811865476f;   // 1/sqrt(2)
  const float i6  = 0.4082482904638631f;   // 1/sqrt(6)
  const float is5 = 0.4472135954999579f;   // 1/sqrt(5)
  // M[i][k] = sum_m C121[i,m,k]*y2[m]
  float M00 = is5 * (-i6 * y2_2 + s2 * y2_4);
  float M01 = is5 * (s2 * y2_0);
  float M02 = is5 * (s2 * y2_3);
  float M11 = is5 * (-i6 * y2_2 - s2 * y2_4);
  float M12 = is5 * (s2 * y2_1);
  float M22 = is5 * (2.0f * i6 * y2_2);
  float M10 = M01, M20 = M02, M21 = M12;

  unsigned short* T = Tbuf + (size_t)e * 11 * 32 + u;   // row stride 32
  T[0 * 32]  = f2bf(sj * y0);
  T[1 * 32]  = f2bf((vj0 * y1_0 + vj1 * y1_1 + vj2 * y1_2) * IS3);
  T[2 * 32]  = f2bf(sj * y1_0 * IS3);
  T[3 * 32]  = f2bf(sj * y1_1 * IS3);
  T[4 * 32]  = f2bf(sj * y1_2 * IS3);
  T[5 * 32]  = f2bf(vj0 * y0 * IS3);
  T[6 * 32]  = f2bf(vj1 * y0 * IS3);
  T[7 * 32]  = f2bf(vj2 * y0 * IS3);
  T[8 * 32]  = f2bf(vj0 * M00 + vj1 * M10 + vj2 * M20);
  T[9 * 32]  = f2bf(vj0 * M01 + vj1 * M11 + vj2 * M21);
  T[10 * 32] = f2bf(vj0 * M02 + vj1 * M12 + vj2 * M22);
}

// ---------------------------------------------------------------------------
// Kernel 3: repack W_fc2 (scales folded in) into per-lane B-fragment order.
//   Bpack[p][kc][nt][lane][i], B 32x16 bf16 layout: n = nt*16 + (lane&15),
//   klocal = (lane<16 ? 0 : 16) + i   (contiguous k order, same as the A fragment)
// ---------------------------------------------------------------------------
__global__ void k_packB(const float* __restrict__ W_fc2,
                        unsigned short* __restrict__ Bpack) {
  int idx = blockIdx.x * blockDim.x + threadIdx.x;   // < 7*65536
  int i    = idx & 15;
  int lane = (idx >> 4) & 31;
  int nt   = (idx >> 9) & 1;
  int kc   = (idx >> 10) & 63;
  int p    = idx >> 16;
  int n  = nt * 16 + (lane & 15);
  int kl = ((lane < 16) ? 0 : 16) + i;
  // fold 1/sqrt(64) (fc2 norm) and A0 / A1 output scales
  float scale = 0.125f * ((p < 4) ? 0.125f : 0.17677669529663687f);
  float v = W_fc2[(size_t)kc * 7168 + p * 1024 + kl * 32 + n] * scale;
  Bpack[idx] = f2bf(v);
}

// ---------------------------------------------------------------------------
// Kernel 4: zero segment-sum buffers
// ---------------------------------------------------------------------------
__global__ void k_zero(float* __restrict__ sums, float* __restrict__ cnt,
                       int nS, int nC) {
  int idx = blockIdx.x * blockDim.x + threadIdx.x;
  if (idx < nS) sums[idx] = 0.f;
  else if (idx < nS + nC) cnt[idx - nS] = 0.f;
}

// ---------------------------------------------------------------------------
// Kernel 5: fused WMMA GEMM.  One wave = 16 edges x 32 outputs, K=2048
// chunked by 32.  A fragment = tfrag (per-tile constant) * h[e,kc] built in
// registers.  B chunks (shared by all 4 waves of the block) are staged
// into a double-buffered LDS region with global_load_async_to_lds_b128,
// synchronized via ASYNCcnt + workgroup barriers.
// ---------------------------------------------------------------------------
template <int NR>
__global__ __launch_bounds__(128)
void k_gemm(const unsigned short* __restrict__ Hbuf,
            const unsigned short* __restrict__ Tbuf,
            const unsigned short* __restrict__ Bpack,
            float* __restrict__ m_ij, int E,
            int pBase, int tr0, int tr1, int tr2,
            int colBase, int colStride) {
  __shared__ __align__(16) unsigned short sH[4][16 * 64];
  __shared__ __align__(32) unsigned short sB[2][NR][1024];  // [buf][l][nt*512+lane*16+i]
  const int lane = threadIdx.x & 31;
  const int wave = threadIdx.x >> 5;
  const int eBase = blockIdx.x * 64 + wave * 16;

  // stage the 16 h-rows (bf16) for this wave into LDS, coalesced dwords
  {
    const unsigned int* hsrc = (const unsigned int*)Hbuf;
    unsigned int* hdst = (unsigned int*)(&sH[wave][0]);
    const int lim = E * 32;  // total dwords in Hbuf
#pragma unroll
    for (int t = 0; t < 16; ++t) {
      int dw = t * 32 + lane;
      int g = eBase * 32 + dw;
      hdst[dw] = hsrc[g < lim ? g : 0];
    }
  }

  // async-copy one K-chunk of B (NR fragments x 2KB) into sB[buf];
  // each wave moves a 512B quarter of each fragment chunk (1 x b128 / lane).
  auto issueB = [&](int kc, int buf) {
#pragma unroll
    for (int l = 0; l < NR; ++l) {
      const int p = pBase + l;
      const unsigned short* gsrc =
          Bpack + (size_t)(p * 64 + kc) * 1024 + wave * 256 + lane * 8;
      unsigned int ldst =
          (unsigned int)(size_t)(&sB[buf][l][wave * 256 + lane * 8]);
      asm volatile("global_load_async_to_lds_b128 %0, %1, off"
                   :: "v"(ldst), "v"(gsrc) : "memory");
    }
  };

  issueB(0, 0);

  const int m = lane & 15;
  int e = eBase + m; if (e >= E) e = E - 1;
  const int kb = (lane < 16) ? 0 : 16;  // A 16x32 bf16 fragment K-base

  const int trows[3] = {tr0, tr1, tr2};
  float tf[NR][16];
#pragma unroll
  for (int l = 0; l < NR; ++l) {
    const unsigned short* tb = Tbuf + ((size_t)e * 11 + trows[l]) * 32;
#pragma unroll
    for (int i = 0; i < 8; ++i) {
      tf[l][i]     = bf2f(tb[kb + i]);        // K = kb..kb+7
      tf[l][8 + i] = bf2f(tb[kb + 8 + i]);    // K = kb+8..kb+15
    }
  }

  v8f acc0 = {0.f, 0.f, 0.f, 0.f, 0.f, 0.f, 0.f, 0.f};
  v8f acc1 = {0.f, 0.f, 0.f, 0.f, 0.f, 0.f, 0.f, 0.f};
  const unsigned short* hrow = &sH[wave][m * 64];

#pragma unroll 1
  for (int kc = 0; kc < 64; ++kc) {
    const int buf = kc & 1;
    // prefetch next chunk into the other buffer (its last readers finished
    // at the end-of-iteration barrier of kc-1)
    if (kc + 1 < 64) issueB(kc + 1, buf ^ 1);
    // in-order ASYNCcnt: allowing NR outstanding means the current buffer's
    // copies (issued one iteration ago) have completed
    if constexpr (NR == 2) asm volatile("s_wait_asynccnt 0x2" ::: "memory");
    else                   asm volatile("s_wait_asynccnt 0x3" ::: "memory");
    __syncthreads();   // current buffer visible to all waves (also covers sH on kc==0)

    float hv = bf2f(hrow[kc]);
#pragma unroll
    for (int l = 0; l < NR; ++l) {
      v16bf a;
#pragma unroll
      for (int i = 0; i < 16; ++i) a[i] = (__bf16)(tf[l][i] * hv);
      v16bf b0 = ((const v16bf*)sB[buf][l])[lane];        // nt = 0
      v16bf b1 = ((const v16bf*)sB[buf][l])[32 + lane];   // nt = 1
      acc0 = __builtin_amdgcn_wmma_f32_16x16x32_bf16(
          false, a, false, b0, (short)0, acc0, false, false);
      acc1 = __builtin_amdgcn_wmma_f32_16x16x32_bf16(
          false, a, false, b1, (short)0, acc1, false, false);
    }
    __syncthreads();   // all waves done reading buf before iter kc+1 overwrites it
  }

  // D layout: lane<16 -> M=v, N=lane ; lane>=16 -> M=v+8, N=lane-16
  const int n = lane & 15;
  const int mOff = (lane < 16) ? 0 : 8;
#pragma unroll
  for (int v = 0; v < 8; ++v) {
    int ee = eBase + mOff + v;
    if (ee < E) {
      size_t row = (size_t)ee * 160 + colBase;
      m_ij[row + (size_t)n * colStride]        = acc0[v];
      m_ij[row + (size_t)(16 + n) * colStride] = acc1[v];
    }
  }
}

// ---------------------------------------------------------------------------
// Kernels 6/7: segment sum via atomics
// ---------------------------------------------------------------------------
__global__ void k_cnt(const int* __restrict__ edge_dst,
                      float* __restrict__ cnt, int E) {
  int e = blockIdx.x * blockDim.x + threadIdx.x;
  if (e < E) atomicAdd(&cnt[edge_dst[e]], 1.0f);
}

__global__ void k_scatter(const float* __restrict__ m_ij,
                          const int* __restrict__ edge_dst,
                          float* __restrict__ sums, int E) {
  int idx = blockIdx.x * blockDim.x + threadIdx.x;
  if (idx >= E * 160) return;
  int e = idx / 160, c = idx - e * 160;
  atomicAdd(&sums[(size_t)edge_dst[e] * 160 + c], m_ij[idx]);
}

// ---------------------------------------------------------------------------
// Kernel 8: gates + self-connection + norms -> out[N][64]
// ---------------------------------------------------------------------------
__global__ void k_final(const float* __restrict__ x,
                        const float* __restrict__ sums,
                        const float* __restrict__ cnt,
                        const float* __restrict__ W_sc_s,
                        const float* __restrict__ W_sc_v,
                        float* __restrict__ out, int N) {
  int idx = blockIdx.x * blockDim.x + threadIdx.x;
  if (idx >= N * 32) return;
  int n = idx >> 5, u = idx & 31;
  float c = cnt[n]; if (c < 1.f) c = 1.f;
  float inv = 1.f / c;
  const float* S = sums + (size_t)n * 160;
  float ms  = S[u] * inv;
  float mg  = S[32 + u] * inv;
  float gv0 = S[64 + u * 3 + 0] * inv;
  float gv1 = S[64 + u * 3 + 1] * inv;
  float gv2 = S[64 + u * 3 + 2] * inv;
  float gs   = ms / (1.f + __expf(-ms));     // silu
  float gate = 1.f / (1.f + __expf(-mg));    // sigmoid
  gv0 *= gate; gv1 *= gate; gv2 *= gate;

  const float* xr = x + (size_t)n * 128;
  const float ism = 0.17677669529663687f;    // 1/sqrt(MUL)
  float scs = 0.f, scv0 = 0.f, scv1 = 0.f, scv2 = 0.f;
#pragma unroll 8
  for (int up = 0; up < 32; ++up) {
    float ws = W_sc_s[up * 32 + u];
    float wv = W_sc_v[up * 32 + u];
    scs  += xr[up] * ws;
    scv0 += xr[32 + up * 3 + 0] * wv;
    scv1 += xr[32 + up * 3 + 1] * wv;
    scv2 += xr[32 + up * 3 + 2] * wv;
  }
  float hs  = gs  + scs  * ism;
  float hv0 = gv0 + scv0 * ism;
  float hv1 = gv1 + scv1 * ism;
  float hv2 = gv2 + scv2 * ism;
  out[(size_t)n * 64 + u]      = sqrtf(hs * hs + 1e-12f);
  out[(size_t)n * 64 + 32 + u] = sqrtf(hv0 * hv0 + hv1 * hv1 + hv2 * hv2 + 1e-12f);
}

// ---------------------------------------------------------------------------
extern "C" void kernel_launch(void* const* d_in, const int* in_sizes, int n_in,
                              void* d_out, int out_size, void* d_ws, size_t ws_size,
                              hipStream_t stream) {
  const float* x           = (const float*)d_in[0];
  const float* edge_attr   = (const float*)d_in[1];
  const float* edge_length = (const float*)d_in[2];
  const int*   edge_src    = (const int*)d_in[3];
  const int*   edge_dst    = (const int*)d_in[4];
  const float* W_fc1       = (const float*)d_in[5];
  const float* W_fc2       = (const float*)d_in[6];
  const float* W_sc_s      = (const float*)d_in[7];
  const float* W_sc_v      = (const float*)d_in[8];
  float* out = (float*)d_out;
  (void)n_in; (void)out_size; (void)ws_size;

  const int N = in_sizes[0] / 128;
  const int E = in_sizes[2];

  char* ws = (char*)d_ws;
  size_t off = 0;
  auto alloc = [&](size_t bytes) -> void* {
    void* p = (void*)(ws + off);
    off += (bytes + 255) & ~(size_t)255;
    return p;
  };
  unsigned short* Hbuf  = (unsigned short*)alloc((size_t)E * 64 * 2);
  unsigned short* Tbuf  = (unsigned short*)alloc((size_t)E * 11 * 32 * 2);
  unsigned short* Bpack = (unsigned short*)alloc((size_t)7 * 65536 * 2);
  float* m_ij = (float*)alloc((size_t)E * 160 * 4);
  float* sums = (float*)alloc((size_t)N * 160 * 4);
  float* cnt  = (float*)alloc((size_t)N * 4);

  k_radial_h<<<(E * 64 + 255) / 256, 256, 0, stream>>>(edge_length, W_fc1, Hbuf, E);
  k_tvec<<<(E * 32 + 255) / 256, 256, 0, stream>>>(x, edge_attr, edge_src, Tbuf, E);
  k_packB<<<(7 * 65536) / 256, 256, 0, stream>>>(W_fc2, Bpack);
  k_zero<<<((N * 160 + N) + 255) / 256, 256, 0, stream>>>(sums, cnt, N * 160, N);

  const int nTiles = (E + 63) / 64;
  // job 0: out_s  = G(t00,p0)+G(t11,p1)        cols [0,32)
  k_gemm<2><<<nTiles, 128, 0, stream>>>(Hbuf, Tbuf, Bpack, m_ij, E, 0, 0, 1, 0,  0, 1);
  // job 1: out_g  = G(t00,p2)+G(t11,p3)        cols [32,64)
  k_gemm<2><<<nTiles, 128, 0, stream>>>(Hbuf, Tbuf, Bpack, m_ij, E, 2, 0, 1, 0, 32, 1);
  // jobs 2-4: out_v[:, :, k], cols 64 + w*3 + k
  k_gemm<3><<<nTiles, 128, 0, stream>>>(Hbuf, Tbuf, Bpack, m_ij, E, 4, 2, 5,  8, 64, 3);
  k_gemm<3><<<nTiles, 128, 0, stream>>>(Hbuf, Tbuf, Bpack, m_ij, E, 4, 3, 6,  9, 65, 3);
  k_gemm<3><<<nTiles, 128, 0, stream>>>(Hbuf, Tbuf, Bpack, m_ij, E, 4, 4, 7, 10, 66, 3);

  k_cnt<<<(E + 255) / 256, 256, 0, stream>>>(edge_dst, cnt, E);
  k_scatter<<<(E * 160 + 255) / 256, 256, 0, stream>>>(m_ij, edge_dst, sums, E);
  k_final<<<(N * 32 + 255) / 256, 256, 0, stream>>>(x, sums, cnt, W_sc_s, W_sc_v, out, N);
}
